// DescrptSeT_403726926075
// MI455X (gfx1250) — matrix operand order repaired and checked by reading, not compiled, not run on hardware
//
#include <hip/hip_runtime.h>
#include <math.h>

typedef __attribute__((ext_vector_type(16))) _Float16 v16h;
typedef __attribute__((ext_vector_type(8)))  float    v8f;

#define NLOC 256
#define NNEI 96
#define NG   96

// Branch-free tanh: prefer CDNA5 hardware v_tanh_f32 (TRANS32, co-executes with WMMA).
__device__ __forceinline__ float fast_tanh(float x) {
#if __has_builtin(__builtin_amdgcn_tanhf)
  return __builtin_amdgcn_tanhf(x);
#else
  // tanh(x) = sign(x) * (1 - t) / (1 + t),  t = 2^(-2*|x|*log2(e))
  float ax = __builtin_fabsf(x);
  float t  = __builtin_amdgcn_exp2f(ax * -2.8853900817779268f);
  float r  = (1.f - t) * __builtin_amdgcn_rcpf(1.f + t);
  return __builtin_copysignf(r, x);
#endif
}

// A-matrix (16x32 f16) K index for slot s = 2*v + half, given lane-half offset (0 or 8).
__device__ __forceinline__ int aK(int s, int hiOff) {
  int v = s >> 1;
  int k = (v < 4) ? (2 * v) : (2 * v + 8);   // v>=4 -> 16 + 2*(v-4)
  return k + (s & 1) + hiOff;
}
// B-matrix (32x16 f16) K index for slot s, given lane-half offset (0 or 16).
__device__ __forceinline__ int bK(int s, int hiOff16) {
  int v = s >> 1;
  return ((v < 4) ? (2 * v) : (2 * v + 8)) + (s & 1) + (hiOff16 >> 1);
}

union AF { v16h v; _Float16 h[16]; };

__global__ __launch_bounds__(256)
void descrpt_se_t_kernel(const int* __restrict__ nlist,
                         const float* __restrict__ coord,
                         const int* __restrict__ atype,
                         const float* __restrict__ meanp,
                         const float* __restrict__ stdp,
                         const float* __restrict__ W1, const float* __restrict__ B1,
                         const float* __restrict__ W2, const float* __restrict__ B2,
                         const float* __restrict__ W3, const float* __restrict__ B3,
                         float* __restrict__ out)
{
  __shared__ float rrl[NNEI][4];      // rr components (dm channels 1..3)
  __shared__ float resv[NG];          // per-atom result accumulator
  __shared__ float h1l[8][16][25];    // per-wave h1 staging (16 rows x 24 ch, padded)
  __shared__ float h2l[8][16][49];    // per-wave h2 staging (16 rows x 48 ch, padded)

  const int ia   = blockIdx.x;
  const int tid  = threadIdx.x;
  const int wid  = tid >> 5;
  const int lane = tid & 31;
  const int n16  = lane & 15;
  const int hiOff   = (lane >= 16) ? 8 : 0;    // A-frag / C-row-half offset
  const int hiOff16 = hiOff * 2;               // B-frag K offset
  const int rbase   = hiOff;                   // C/D rows this lane holds

  if (tid < NG) resv[tid] = 0.f;

  // ---- env matrix: rr[t] = ((diff/len^2)*sw - mean[1..3]) / std[1..3] ----
  if (tid < NNEI) {
    int t = tid;
    int j = nlist[ia * NNEI + t];
    int msk = (j >= 0) ? 1 : 0;
    int jc = msk ? j : 0;
    float dx = coord[jc * 3 + 0] - coord[ia * 3 + 0];
    float dy = coord[jc * 3 + 1] - coord[ia * 3 + 1];
    float dz = coord[jc * 3 + 2] - coord[ia * 3 + 2];
    float len = sqrtf(dx * dx + dy * dy + dz * dz) + (msk ? 0.f : 1.f);
    float inv2 = 1.f / (len * len);
    const float rmin = 0.5f, rmax = 6.0f;
    float uu = (len - rmin) / (rmax - rmin);
    float vv = uu * uu * uu * (-6.f * uu * uu + 15.f * uu - 10.f) + 1.f;
    float sw = (len <= rmin) ? 1.f : ((len >= rmax) ? 0.f : vv);
    sw *= (float)msk;
    int at = atype[ia];
    const float* mn = meanp + ((at * NNEI + t) * 4);
    const float* sd = stdp  + ((at * NNEI + t) * 4);
    rrl[t][0] = (dx * inv2 * sw - mn[1]) / sd[1];
    rrl[t][1] = (dy * inv2 * sw - mn[2]) / sd[2];
    rrl[t][2] = (dz * inv2 * sw - mn[3]) / sd[3];
    rrl[t][3] = 0.f;
  }
  __syncthreads();

  const int tiv[3] = {0, 0, 1};
  const int tjv[3] = {0, 1, 1};

  for (int p = 0; p < 3; ++p) {
    const int ti = tiv[p], tj = tjv[p];
    const int Si = ti ? 64 : 32;
    const int Sj = tj ? 64 : 32;
    const int bi = ti ? 32 : 0;
    const int bj = tj ? 32 : 0;
    const int sjShift = tj ? 6 : 5;
    const int sjMask  = Sj - 1;
    const float scale = 1.f / (float)(Si * Sj);
    const int ntiles = (Si * Sj) >> 4;   // 64 / 128 / 256, all divisible by 8 waves

    // ---- per-pair weight fragments (held in VGPRs; weights are L2-hot) ----
    float w1v[16], b1v[16];
    AF b2f[3]; float b2v[3];
    AF b3f[6][2]; float b3v[6];
#pragma unroll
    for (int s = 0; s < 16; ++s) {
      int k = aK(s, hiOff);
      w1v[s] = (k < 24) ? W1[p * 24 + k] : 0.f;
      b1v[s] = (k < 24) ? B1[p * 24 + k] : 0.f;
    }
#pragma unroll
    for (int ct = 0; ct < 3; ++ct) {
      int col = ct * 16 + n16;
      b2v[ct] = B2[p * 48 + col];
#pragma unroll
      for (int s = 0; s < 16; ++s) {
        int k = bK(s, hiOff16);
        b2f[ct].h[s] = (_Float16)((k < 24) ? W2[(p * 24 + k) * 48 + col] : 0.f);
      }
    }
#pragma unroll
    for (int ct = 0; ct < 6; ++ct) {
      int col = ct * 16 + n16;
      b3v[ct] = B3[p * 96 + col];
#pragma unroll
      for (int q = 0; q < 2; ++q) {
#pragma unroll
        for (int s = 0; s < 16; ++s) {
          int k = bK(s, hiOff16) + 32 * q;
          b3f[ct][q].h[s] = (_Float16)((k < 48) ? W3[(p * 48 + k) * 96 + col] : 0.f);
        }
      }
    }

    float acc[6] = {0.f, 0.f, 0.f, 0.f, 0.f, 0.f};

    for (int tile = wid; tile < ntiles; tile += 8) {
      // env value for this lane's matrix row (lanes L and L+16 share a row)
      int idx = tile * 16 + n16;
      int ji = idx >> sjShift;
      int jk = idx & sjMask;
      float e = rrl[bi + ji][0] * rrl[bj + jk][0]
              + rrl[bi + ji][1] * rrl[bj + jk][1]
              + rrl[bi + ji][2] * rrl[bj + jk][2];
      float ev[8];
#pragma unroll
      for (int r = 0; r < 8; ++r) ev[r] = __shfl(e, rbase + r, 32);

      // ---- layer 1: h1 = tanh(e*W1 + b1); built straight into the A fragment ----
      AF a2;
#pragma unroll
      for (int s = 0; s < 16; ++s) {
        float h = fast_tanh(e * w1v[s] + b1v[s]);   // pad channels: tanh(0)=0
        a2.h[s] = (_Float16)h;
        int k = aK(s, hiOff);
        if (k < 24) h1l[wid][n16][k] = h;
      }
      asm volatile("s_wait_dscnt 0" ::: "memory");   // wave-private staging ready

      // ---- layer 2: h2 = tanh(h1@W2 + b2) + [h1,h1]  (3 col tiles) ----
#pragma unroll
      for (int ct = 0; ct < 3; ++ct) {
        v8f c;
#pragma unroll
        for (int r = 0; r < 8; ++r) c[r] = b2v[ct];
        v8f d = __builtin_amdgcn_wmma_f32_16x16x32_f16(false, a2.v, false, b2f[ct].v,
                                                       (short)0, c, false, false);
        int col = ct * 16 + n16;
        int c24 = col % 24;
#pragma unroll
        for (int r = 0; r < 8; ++r) {
          float h2 = fast_tanh(d[r]) + h1l[wid][rbase + r][c24];
          h2l[wid][rbase + r][col] = h2;
        }
      }
      asm volatile("s_wait_dscnt 0" ::: "memory");

      // ---- build A fragments for layer 3 (16x64, two K=32 chunks) ----
      AF a3[2];
#pragma unroll
      for (int q = 0; q < 2; ++q) {
#pragma unroll
        for (int s = 0; s < 16; ++s) {
          int k = aK(s, hiOff) + 32 * q;
          a3[q].h[s] = (_Float16)((k < 48) ? h2l[wid][n16][k] : 0.f);
        }
      }

      // ---- layer 3 + env-weighted contraction (6 col tiles x 2 K-chunks) ----
#pragma unroll
      for (int ct = 0; ct < 6; ++ct) {
        v8f c;
#pragma unroll
        for (int r = 0; r < 8; ++r) c[r] = b3v[ct];
        v8f d = __builtin_amdgcn_wmma_f32_16x16x32_f16(false, a3[0].v, false, b3f[ct][0].v,
                                                       (short)0, c, false, false);
        d = __builtin_amdgcn_wmma_f32_16x16x32_f16(false, a3[1].v, false, b3f[ct][1].v,
                                                   (short)0, d, false, false);
        int col = ct * 16 + n16;
        int c48 = col % 48;
#pragma unroll
        for (int r = 0; r < 8; ++r) {
          float h3 = fast_tanh(d[r]) + h2l[wid][rbase + r][c48];
          acc[ct] += h3 * ev[r];
        }
      }
    }

    // flush this pair's contribution (lane halves + 8 waves combine via LDS atomics)
#pragma unroll
    for (int ct = 0; ct < 6; ++ct)
      atomicAdd(&resv[ct * 16 + n16], acc[ct] * scale);
  }

  __syncthreads();
  if (tid < NG) out[ia * NG + tid] = resv[tid];
}

extern "C" void kernel_launch(void* const* d_in, const int* in_sizes, int n_in,
                              void* d_out, int out_size, void* d_ws, size_t ws_size,
                              hipStream_t stream) {
  (void)in_sizes; (void)n_in; (void)out_size; (void)d_ws; (void)ws_size;
  descrpt_se_t_kernel<<<NLOC, 256, 0, stream>>>(
      (const int*)d_in[0],    // nlist
      (const float*)d_in[1],  // extended_coord
      (const int*)d_in[2],    // extended_atype
      (const float*)d_in[3],  // mean
      (const float*)d_in[4],  // stddev
      (const float*)d_in[5], (const float*)d_in[6],   // W1, b1
      (const float*)d_in[7], (const float*)d_in[8],   // W2, b2
      (const float*)d_in[9], (const float*)d_in[10],  // W3, b3
      (float*)d_out);
}
